// LinearKernel_26293789786376
// MI455X (gfx1250) — matrix operand order repaired and checked by reading, not compiled, not run on hardware
//
#include <hip/hip_runtime.h>

typedef __attribute__((ext_vector_type(16))) _Float16 v16h;
typedef __attribute__((ext_vector_type(8)))  float    v8f;

#define WAVES 8
#define BLOCK (WAVES * 32)
#define KSTR  17    // padded stride (floats) for kernel_flat staging

__device__ __forceinline__ float gelu_tanh(float v) {
    // jax.nn.gelu (approximate=True): 0.5*x*(1+tanh(u)),
    // u = sqrt(2/pi)*(x + 0.044715 x^3) = x*(a + b*x^2),  b = a*0.044715
#if __has_builtin(__builtin_amdgcn_tanhf)
    float v2 = v * v;
    float u  = v * fmaf(v2, 0.035677408136300125f, 0.7978845608028654f);
    float t  = __builtin_amdgcn_tanhf(u);         // v_tanh_f32
    float xh = 0.5f * v;
    return fmaf(xh, t, xh);                       // 0.5x + 0.5x*t
#else
    float v2 = v * v;
    float u2 = -2.0f * v * fmaf(v2, 0.035677408136300125f, 0.7978845608028654f);
    float s  = __builtin_amdgcn_rcpf(1.0f + __expf(u2));
    return v * s;                                 // x * sigmoid(2u)
#endif
}

__global__ __launch_bounds__(BLOCK) void equiv_ff_kernel(
    const float* __restrict__ x,      // [E,4,3]
    const float* __restrict__ feats,  // [E,32]
    const float* __restrict__ W1,     // [32,256]
    const float* __restrict__ b1,     // [256]
    const float* __restrict__ W2,     // [256,16]
    const float* __restrict__ b2,     // [16]
    float* __restrict__ out,          // [E,4,3]
    int E, int NT)
{
    // GEMM1 is computed transposed:  h^T_tile = (W1^T tile) x feats^T
    // sW1: W1^T n-tiles in WMMA *A* layout  (lane: row n_local=l&15, K chunks {8h..,16+8h..})
    // sW2: W2 k-chunks   in WMMA *B* layout  (lane: col n2=l&15, K chunks {8h..,16+8h..})
    __shared__ v16h sW1[16 * 32];                     // 16 KB
    __shared__ v16h sW2[8 * 32];                      //  8 KB
    __shared__ __align__(16) float sB1[256];          //  1 KB
    __shared__ __align__(16) float kbuf[WAVES][16][KSTR];

    const int tid = threadIdx.x;

    // ---- one-time per block: swizzle f32 weights -> f16 WMMA layouts in LDS ----
    for (int rec = tid; rec < 16 * 32; rec += BLOCK) {
        int t  = rec >> 5, l = rec & 31;
        int nn = 16 * t + (l & 15);
        int h2 = l >> 4;
        v16h v;
        #pragma unroll
        for (int q = 0; q < 8; ++q)
            v[q] = (_Float16)W1[(size_t)(8 * h2 + q) * 256 + nn];          // K = 8h+q
        #pragma unroll
        for (int q = 0; q < 8; ++q)
            v[8 + q] = (_Float16)W1[(size_t)(16 + 8 * h2 + q) * 256 + nn]; // K = 16+8h+q
        sW1[rec] = v;
    }
    for (int rec = tid; rec < 8 * 32; rec += BLOCK) {
        int s  = rec >> 5, l = rec & 31;
        int n  = l & 15;
        int k0 = 32 * s + 8 * (l >> 4);
        v16h v;
        #pragma unroll
        for (int q = 0; q < 16; ++q)
            v[q] = (_Float16)W2[(size_t)(k0 + q + (q & 8)) * 16 + n];
        sW2[rec] = v;
    }
    for (int i = tid; i < 256; i += BLOCK) sB1[i] = b1[i];
    __syncthreads();

    const int wv   = tid >> 5;
    const int lane = tid & 31;
    const int m    = lane & 15;   // edge-in-tile (and n2 column id)
    const int hh   = lane >> 4;   // half-wave

    // b2 seeds the GEMM2 accumulator: C2[r] layout is [M=edge][N=l&15] -> lane-uniform
    const float bias2 = b2[m];
    v8f cb2;
    #pragma unroll
    for (int r = 0; r < 8; ++r) cb2[r] = bias2;

    const int wstride = (int)gridDim.x * WAVES;

    for (int tile = blockIdx.x * WAVES + wv; tile < NT; tile += wstride) {
        const long e0 = (long)tile * 16;
        long em = e0 + m; if (em >= E) em = E - 1;   // clamp loads on ragged tail

        // ---- B (feats^T 32x16): lane = col edge m, K = 8h..8h+7, 16+8h..16+8h+7 ----
        const float* rp = feats + em * 32 + 8 * hh;
        float4 f0 = *(const float4*)(rp);
        float4 f1 = *(const float4*)(rp + 4);
        float4 f2 = *(const float4*)(rp + 16);
        float4 f3 = *(const float4*)(rp + 20);
        v16h bf;
        bf[0]  = (_Float16)f0.x; bf[1]  = (_Float16)f0.y; bf[2]  = (_Float16)f0.z; bf[3]  = (_Float16)f0.w;
        bf[4]  = (_Float16)f1.x; bf[5]  = (_Float16)f1.y; bf[6]  = (_Float16)f1.z; bf[7]  = (_Float16)f1.w;
        bf[8]  = (_Float16)f2.x; bf[9]  = (_Float16)f2.y; bf[10] = (_Float16)f2.z; bf[11] = (_Float16)f2.w;
        bf[12] = (_Float16)f3.x; bf[13] = (_Float16)f3.y; bf[14] = (_Float16)f3.z; bf[15] = (_Float16)f3.w;

        // prefetch next tile's inputs (global_prefetch_b8)
        long en = e0 + (long)wstride * 16 + m;
        if (en < E) {
            __builtin_prefetch(feats + en * 32, 0, 1);
            __builtin_prefetch(x + en * 12, 0, 1);
        }

        // ---- fused GEMM1 (transposed, bias via C operand) -> GELU -> GEMM2 ----
        v8f c2;
        #pragma unroll
        for (int s = 0; s < 8; ++s) {
            float g[2][8];
            #pragma unroll
            for (int half = 0; half < 2; ++half) {
                int t = 2 * s + half;
                // C = b1 broadcast: C[r] = b1[16t + 8h + r]  (depends on M only)
                union { v8f v; uint4 q[2]; } cb;
                cb.q[0] = *(const uint4*)&sB1[16 * t + 8 * hh];
                cb.q[1] = *(const uint4*)&sB1[16 * t + 8 * hh + 4];
                v8f c = __builtin_amdgcn_wmma_f32_16x16x32_f16(
                    false, sW1[t * 32 + lane], false, bf, (short)0, cb.v, false, false);
                // c[r] = h[edge=m][n = 16t + 8h + r] + b1[n]
                #pragma unroll
                for (int r = 0; r < 8; ++r) g[half][r] = gelu_tanh(c[r]);
            }
            // pack straight into GEMM2 A layout: elems 0..7 = K 8h+e, 8..15 = K 16+8h+e
            v16h a2;
            #pragma unroll
            for (int q = 0; q < 8; ++q) {
                a2[q]     = (_Float16)g[0][q];
                a2[8 + q] = (_Float16)g[1][q];
            }
            c2 = __builtin_amdgcn_wmma_f32_16x16x32_f16(
                false, a2, false, sW2[s * 32 + lane], (short)0,
                (s == 0) ? cb2 : c2, false, false);
        }

        // c2[r] = kernel_flat[edge = r+8h][n2 = m] + b2[n2]
        #pragma unroll
        for (int r = 0; r < 8; ++r)
            kbuf[wv][r + 8 * hh][m] = c2[r];

        // ---- einsum('ij,jk->ik'): 2 lanes per edge (lane hh owns rows 2hh,2hh+1) ----
        const float* xp = x + em * 12;
        float4 x0 = *(const float4*)(xp);
        float4 x1 = *(const float4*)(xp + 4);
        float4 x2 = *(const float4*)(xp + 8);
        // x[j][s]: {x0.x,x0.y,x0.z},{x0.w,x1.x,x1.y},{x1.z,x1.w,x2.x},{x2.y,x2.z,x2.w}
        float kv[8];
        #pragma unroll
        for (int q = 0; q < 8; ++q) kv[q] = kbuf[wv][m][8 * hh + q]; // n2 = 8hh+q = i*4+j

        if (e0 + m < E) {
            float* op = out + (e0 + m) * 12 + 6 * hh;   // rows i = 2hh, 2hh+1
            #pragma unroll
            for (int q2 = 0; q2 < 2; ++q2) {
                float k0v = kv[4 * q2 + 0], k1v = kv[4 * q2 + 1];
                float k2v = kv[4 * q2 + 2], k3v = kv[4 * q2 + 3];
                float o0 = k0v * x0.x + k1v * x0.w + k2v * x1.z + k3v * x2.y;
                float o1 = k0v * x0.y + k1v * x1.x + k2v * x1.w + k3v * x2.z;
                float o2 = k0v * x0.z + k1v * x1.y + k2v * x2.x + k3v * x2.w;
                op[q2 * 3 + 0] = o0;
                op[q2 * 3 + 1] = o1;
                op[q2 * 3 + 2] = o2;
            }
        }
    }
}

extern "C" void kernel_launch(void* const* d_in, const int* in_sizes, int n_in,
                              void* d_out, int out_size, void* d_ws, size_t ws_size,
                              hipStream_t stream) {
    const float* x  = (const float*)d_in[0];
    const float* f  = (const float*)d_in[1];
    const float* W1 = (const float*)d_in[2];
    const float* b1 = (const float*)d_in[3];
    const float* W2 = (const float*)d_in[4];
    const float* b2 = (const float*)d_in[5];
    float* out = (float*)d_out;

    const int E  = in_sizes[1] / 32;          // inv_feats is [E,32]
    const int NT = (E + 15) / 16;             // 16-edge tiles

    int blocks = (NT + WAVES - 1) / WAVES;
    if (blocks > 8192) blocks = 8192;         // grid-stride persistent-ish
    if (blocks < 1) blocks = 1;

    hipLaunchKernelGGL(equiv_ff_kernel, dim3(blocks), dim3(BLOCK), 0, stream,
                       x, f, W1, b1, W2, b2, out, E, NT);
}
